// EdgeConvBlock_25623774888365
// MI455X (gfx1250) — matrix operand order repaired and checked by reading, not compiled, not run on hardware
//
#include <hip/hip_runtime.h>
#include <hip/hip_bf16.h>
#include <cstdint>

// ---------------------------------------------------------------------------
// EdgeConv block on MI455X (gfx1250, wave32, WMMA f32_16x16x32_f16)
//   N=30000 points, K=16 neighbors, C_IN=128, C_OUT=256
//   edge(16x256) @ W1(256x256) -> gelu -> @ W2(256x256) -> max_k -> +skip -> LN
//
// v2: batch PBATCH=4 points per workgroup. Each wave owns 2 of 16 output
// tiles; per tile it loads the 8 B-fragments (64 VGPRs) ONCE from L2 and
// reuses them across the 4 points (4x L2 weight-traffic reduction:
// 7.7 GB -> 1.9 GB). Skip GEMV restructured to read Ws once per block.
// ---------------------------------------------------------------------------

typedef _Float16 half8  __attribute__((ext_vector_type(8)));
typedef _Float16 half16 __attribute__((ext_vector_type(16)));
typedef float    float8 __attribute__((ext_vector_type(8)));

#define C_IN   128
#define C_OUT  256
#define KNB    16
#define LDE    264   // LDS row stride (halfs); 528B breaks the 64-bank alignment
#define PBATCH 4

// A-matrix fragment (16x32 f16) from row-major LDS tile (contiguous K, matches B):
//   lane<16 : M=lane,    elems 0..15 -> K=k0+0..15
//   lane>=16: M=lane-16, elems 0..15 -> K=k0+16..31
__device__ inline half16 load_a_frag(const _Float16* base, int k0, int lane) {
    int m   = lane & 15;
    int klo = (lane & 16) ? 16 : 0;
    const _Float16* p = base + m * LDE + k0 + klo;
    union { half16 v; half8 h[2]; } u;
    u.h[0] = *(const half8*)(p);
    u.h[1] = *(const half8*)(p + 8);
    return u.v;
}

// B-matrix fragment (32x16 f16) from transposed (out-channel-major) f16 weights:
//   lane<16 : N=lane, K=k0+0..15 ; lane>=16: N=lane-16, K=k0+16..31
__device__ inline half16 load_b_frag(const _Float16* __restrict__ wT, int tile, int k0, int lane) {
    int n  = lane & 15;
    int kb = (lane & 16) ? 16 : 0;
    const _Float16* p = wT + (size_t)(tile * 16 + n) * C_OUT + k0 + kb;
    return *(const half16*)p;
}

__device__ inline float gelu_exact(float x) {
    return 0.5f * x * (1.0f + erff(x * 0.70710678118654752f));
}

// Transpose + f32->f16 convert both weight matrices into workspace (one-time).
__global__ void convert_weights(const float* __restrict__ W1, const float* __restrict__ W2,
                                _Float16* __restrict__ W1T, _Float16* __restrict__ W2T) {
    int e = blockIdx.x;    // output channel 0..255
    int c = threadIdx.x;   // input  channel 0..255
    W1T[e * C_OUT + c] = (_Float16)W1[c * C_OUT + e];
    W2T[e * C_OUT + c] = (_Float16)W2[c * C_OUT + e];
}

__global__ __launch_bounds__(256)
void edgeconv_main(const float* __restrict__ feat, const int* __restrict__ knn,
                   const _Float16* __restrict__ W1T, const float* __restrict__ b1,
                   const _Float16* __restrict__ W2T, const float* __restrict__ b2,
                   const float* __restrict__ Ws,  const float* __restrict__ bs,
                   const float* __restrict__ gamma, const float* __restrict__ beta,
                   float* __restrict__ out, int N) {
    __shared__ __align__(16) _Float16 edgeA[PBATCH][16 * LDE]; // edge f16, 16x256 per point
    __shared__ __align__(16) _Float16 hA[PBATCH][16 * LDE];    // gelu(edge@W1+b1) f16
    __shared__ __align__(16) float featC[PBATCH][C_IN];        // center features f32
    __shared__ float vals[PBATCH][C_OUT];                      // max_k result
    __shared__ float red[PBATCH][2];                           // sum, sumsq per point

    const int n0   = blockIdx.x * PBATCH;
    const int t    = threadIdx.x;
    const int lane = t & 31;
    const int wave = t >> 5;

    if (t < 2 * PBATCH) red[t >> 1][t & 1] = 0.0f;

    // Stage center features: 128 threads, one float4 each (p = t>>5, q = t&31).
    if (t < 32 * PBATCH) {
        int p = t >> 5, q = t & 31;
        int np = min(n0 + p, N - 1);
        ((float4*)featC[p])[q] = ((const float4*)(feat + (size_t)np * C_IN))[q];
    }
    __syncthreads();

    // Build edge matrices: row j = neighbor, cols [0,128)=center, [128,256)=nbr-center.
    {
        const int j  = t >> 4;         // neighbor 0..15
        const int cb = (t & 15) * 8;   // channel block
#pragma unroll
        for (int p = 0; p < PBATCH; ++p) {
            int np = min(n0 + p, N - 1);
            int nb = knn[np * KNB + j];
            const float* cp  = featC[p] + cb;
            const float* nbp = feat + (size_t)nb * C_IN + cb;
#pragma unroll
            for (int i = 0; i < 8; ++i) {
                float c = cp[i];
                float v = nbp[i];
                edgeA[p][j * LDE + cb + i]        = (_Float16)c;
                edgeA[p][j * LDE + C_IN + cb + i] = (_Float16)(v - c);
            }
        }
    }
    __syncthreads();

    // ---- GEMM1: h = gelu(edge @ W1 + b1). B fragments register-resident. ----
#pragma unroll
    for (int tt = 0; tt < 2; ++tt) {
        const int tile = wave + tt * 8;
        half16 bf[8];
#pragma unroll
        for (int ks = 0; ks < 8; ++ks) bf[ks] = load_b_frag(W1T, tile, ks * 32, lane);

        float8 accs[PBATCH];
#pragma unroll
        for (int p = 0; p < PBATCH; ++p) {
            float8 acc = {};
#pragma unroll
            for (int ks = 0; ks < 8; ++ks) {
                half16 a = load_a_frag(edgeA[p], ks * 32, lane);
                acc = __builtin_amdgcn_wmma_f32_16x16x32_f16(
                    false, a, false, bf[ks], (short)0, acc, false, false);
            }
            accs[p] = acc;
        }

        const int col  = tile * 16 + (lane & 15);
        const int mb   = (lane & 16) ? 8 : 0;
        const float bias = b1[col];
#pragma unroll
        for (int p = 0; p < PBATCH; ++p) {
#pragma unroll
            for (int r = 0; r < 8; ++r) {
                float x = accs[p][r] + bias;
                hA[p][(mb + r) * LDE + col] = (_Float16)gelu_exact(x);
            }
        }
    }
    __syncthreads();

    // ---- GEMM2 + bias + max over K ----
#pragma unroll
    for (int tt = 0; tt < 2; ++tt) {
        const int tile = wave + tt * 8;
        half16 bf[8];
#pragma unroll
        for (int ks = 0; ks < 8; ++ks) bf[ks] = load_b_frag(W2T, tile, ks * 32, lane);

        float8 accs[PBATCH];
#pragma unroll
        for (int p = 0; p < PBATCH; ++p) {
            float8 acc = {};
#pragma unroll
            for (int ks = 0; ks < 8; ++ks) {
                half16 a = load_a_frag(hA[p], ks * 32, lane);
                acc = __builtin_amdgcn_wmma_f32_16x16x32_f16(
                    false, a, false, bf[ks], (short)0, acc, false, false);
            }
            accs[p] = acc;
        }

        const int col  = tile * 16 + (lane & 15);
        const float bias = b2[col];
#pragma unroll
        for (int p = 0; p < PBATCH; ++p) {
            float m = accs[p][0] + bias;
#pragma unroll
            for (int r = 1; r < 8; ++r) m = fmaxf(m, accs[p][r] + bias);
            // lane l holds rows {mb..mb+7}; lane l^16 has the other 8 rows of this col
            m = fmaxf(m, __shfl_xor(m, 16, 32));
            if (lane < 16) vals[p][col] = m;
        }
    }
    __syncthreads();

    // ---- Skip GEMV (Ws read once per block, coalesced) + LayerNorm ----
    float sk[PBATCH];
#pragma unroll
    for (int p = 0; p < PBATCH; ++p) sk[p] = bs[t];
#pragma unroll 8
    for (int c = 0; c < C_IN; ++c) {
        float w = Ws[c * C_OUT + t];
#pragma unroll
        for (int p = 0; p < PBATCH; ++p) sk[p] += featC[p][c] * w;
    }

    float v[PBATCH];
#pragma unroll
    for (int p = 0; p < PBATCH; ++p) {
        v[p] = vals[p][t] + sk[p];
        float s1 = v[p], s2 = v[p] * v[p];
#pragma unroll
        for (int off = 16; off > 0; off >>= 1) {
            s1 += __shfl_xor(s1, off, 32);
            s2 += __shfl_xor(s2, off, 32);
        }
        if (lane == 0) {
            atomicAdd(&red[p][0], s1);
            atomicAdd(&red[p][1], s2);
        }
    }
    __syncthreads();

    const float g = gamma[t], bt = beta[t];
#pragma unroll
    for (int p = 0; p < PBATCH; ++p) {
        if (n0 + p < N) {
            float mu   = red[p][0] * (1.0f / 256.0f);
            float var  = red[p][1] * (1.0f / 256.0f) - mu * mu;
            float rstd = rsqrtf(var + 1e-5f);
            out[(size_t)(n0 + p) * C_OUT + t] = (v[p] - mu) * rstd * g + bt;
        }
    }
}

extern "C" void kernel_launch(void* const* d_in, const int* in_sizes, int n_in,
                              void* d_out, int out_size, void* d_ws, size_t ws_size,
                              hipStream_t stream) {
    const float* feat  = (const float*)d_in[0];
    const int*   knn   = (const int*)  d_in[1];   // knn_idx (canonicalized int32)
    const float* W1    = (const float*)d_in[2];
    const float* b1    = (const float*)d_in[3];
    const float* W2    = (const float*)d_in[4];
    const float* b2    = (const float*)d_in[5];
    const float* Ws    = (const float*)d_in[6];
    const float* bs    = (const float*)d_in[7];
    const float* gamma = (const float*)d_in[8];
    const float* beta  = (const float*)d_in[9];

    _Float16* W1T = (_Float16*)d_ws;
    _Float16* W2T = (_Float16*)((char*)d_ws + (size_t)C_OUT * C_OUT * sizeof(_Float16));

    const int N = in_sizes[0] / C_IN;  // 30000
    const int nblk = (N + PBATCH - 1) / PBATCH;

    convert_weights<<<C_OUT, C_OUT, 0, stream>>>(W1, W2, W1T, W2T);
    edgeconv_main<<<nblk, 256, 0, stream>>>(feat, knn, W1T, b1, W2T, b2,
                                            Ws, bs, gamma, beta, (float*)d_out, N);
}
